// AttenMlpFinal_85418309583437
// MI455X (gfx1250) — matrix operand order repaired and checked by reading, not compiled, not run on hardware
//
#include <hip/hip_runtime.h>

typedef __attribute__((ext_vector_type(16))) _Float16 v16h;
typedef __attribute__((ext_vector_type(8)))  _Float16 v8h;
typedef __attribute__((ext_vector_type(8)))  float    v8f;

#define BSZ   32768
#define MHOP  8
#define DDIM  128
#define HDIM  512
#define KDIM  256                 // 2*D
#define W1_STRIDE 264             // padded halves per W1 row (conflict-free B reads)
#define LDS_BYTES (HDIM * W1_STRIDE * 2)   // 270336 bytes

__global__ __launch_bounds__(256, 1)
void atten_mlp_kernel(const float* __restrict__ q,
                      const float* __restrict__ k,
                      const float* __restrict__ v,
                      const float* __restrict__ W1,
                      const float* __restrict__ W2,
                      float* __restrict__ out)
{
    extern __shared__ _Float16 w1h[];

    const int tid = threadIdx.x;

    // ---- stage W1 (f32 -> f16) into padded LDS, cooperatively per block ----
    for (int i = tid; i < HDIM * (KDIM / 4); i += 256) {
        const int row = i >> 6;          // 0..511
        const int c4  = i & 63;          // group of 4 columns
        float4 f = reinterpret_cast<const float4*>(W1)[row * 64 + c4];
        union { _Float16 h[4]; uint32_t u[2]; } pk;
        pk.h[0] = (_Float16)f.x; pk.h[1] = (_Float16)f.y;
        pk.h[2] = (_Float16)f.z; pk.h[3] = (_Float16)f.w;
        uint32_t* dst = reinterpret_cast<uint32_t*>(&w1h[row * W1_STRIDE + c4 * 4]);
        dst[0] = pk.u[0];
        dst[1] = pk.u[1];
    }
    __syncthreads();

    const int wave  = tid >> 5;
    const int lane  = tid & 31;
    const int nlan  = lane & 15;       // row (A) / column (B,D) index within tile
    const int khalf = lane >> 4;       // which K half this lane carries
    const int b0    = blockIdx.x * 32 + wave * 4;   // 4 batches per wave (2 tiles)

    // ---- build resident A fragments (ISA 16-bit A layout) for both tiles ----
    v16h A[2][8];
    const int roff = khalf * 8;        // lanes 16-31 carry K offsets +8 / +24
    #pragma unroll
    for (int t = 0; t < 2; ++t) {
        const int bb = b0 + t * 2 + (nlan >> 3);   // rows 0-7 -> batch, 8-15 -> batch+1
        const int mm = nlan & 7;
        const float* qrow = q + bb * DDIM;
        const float* krow = k + (bb * MHOP + mm) * DDIM;
        #pragma unroll
        for (int kc = 0; kc < 8; ++kc) {
            const int c1 = kc * 32 + roff;         // xx column of first 8-run
            const int c2 = c1 + 16;                // second 8-run
            const float* s1 = (kc < 4) ? (qrow + c1) : (krow + (c1 - DDIM));
            const float* s2 = (kc < 4) ? (qrow + c2) : (krow + (c2 - DDIM));
            float4 f0 = reinterpret_cast<const float4*>(s1)[0];
            float4 f1 = reinterpret_cast<const float4*>(s1)[1];
            float4 f2 = reinterpret_cast<const float4*>(s2)[0];
            float4 f3 = reinterpret_cast<const float4*>(s2)[1];
            v16h a;
            a[0] = (_Float16)f0.x; a[1] = (_Float16)f0.y; a[2]  = (_Float16)f0.z; a[3]  = (_Float16)f0.w;
            a[4] = (_Float16)f1.x; a[5] = (_Float16)f1.y; a[6]  = (_Float16)f1.z; a[7]  = (_Float16)f1.w;
            a[8] = (_Float16)f2.x; a[9] = (_Float16)f2.y; a[10] = (_Float16)f2.z; a[11] = (_Float16)f2.w;
            a[12]= (_Float16)f3.x; a[13]= (_Float16)f3.y; a[14] = (_Float16)f3.z; a[15] = (_Float16)f3.w;
            A[t][kc] = a;
        }
    }

    // ---- GEMM over hidden dim; fuse relu + w2 dot into score accumulation ----
    float sacc[2][8];
    #pragma unroll
    for (int t = 0; t < 2; ++t)
        #pragma unroll
        for (int i = 0; i < 8; ++i) sacc[t][i] = 0.f;

    const _Float16* bbase = w1h + nlan * W1_STRIDE + khalf * 8;
    #pragma unroll 1
    for (int nt = 0; nt < 32; ++nt) {
        const int n0 = nt * 16;
        const _Float16* bp = bbase + n0 * W1_STRIDE;
        v8f c0 = {};
        v8f c1 = {};
        #pragma unroll
        for (int kc = 0; kc < 8; ++kc) {
            // B fragment: lane (n = nlan) holds K = kc*32 + khalf*8 + 0..7 and kc*32 + 16 + khalf*8 + 0..7
            v8h blo = *reinterpret_cast<const v8h*>(bp + kc * 32);
            v8h bhi = *reinterpret_cast<const v8h*>(bp + kc * 32 + 16);
            v16h B;
            #pragma unroll
            for (int j = 0; j < 8; ++j) { B[j] = blo[j]; B[8 + j] = bhi[j]; }
            c0 = __builtin_amdgcn_wmma_f32_16x16x32_f16(false, A[0][kc], false, B,
                                                        (short)0, c0, false, false);
            c1 = __builtin_amdgcn_wmma_f32_16x16x32_f16(false, A[1][kc], false, B,
                                                        (short)0, c1, false, false);
        }
        const float w2v = W2[n0 + nlan];
        #pragma unroll
        for (int i = 0; i < 8; ++i) {
            float h0 = c0[i]; h0 = h0 > 0.f ? h0 : 0.f;
            float h1 = c1[i]; h1 = h1 > 0.f ? h1 : 0.f;
            sacc[0][i] += h0 * w2v;
            sacc[1][i] += h1 * w2v;
        }
    }

    // ---- per-tile: reduce scores across 16-lane halves, softmax, weighted V sum ----
    #pragma unroll
    for (int t = 0; t < 2; ++t) {
        float att[8];
        float mx = -1e30f;
        #pragma unroll
        for (int i = 0; i < 8; ++i) {
            float s = sacc[t][i];
            s += __shfl_xor(s, 1, 32);
            s += __shfl_xor(s, 2, 32);
            s += __shfl_xor(s, 4, 32);
            s += __shfl_xor(s, 8, 32);   // full row sum within each 16-lane half
            att[i] = s;
            mx = fmaxf(mx, s);
        }
        float sum = 0.f;
        #pragma unroll
        for (int i = 0; i < 8; ++i) { att[i] = __expf(att[i] - mx); sum += att[i]; }
        const float inv = 1.f / sum;

        const int ob = b0 + t * 2 + khalf;   // lanes 0-15: even batch, 16-31: odd batch
        const int d0 = nlan * 8;             // 8 output dims per lane
        float acc[8] = {0.f, 0.f, 0.f, 0.f, 0.f, 0.f, 0.f, 0.f};
        #pragma unroll
        for (int m = 0; m < 8; ++m) {
            const float w = att[m] * inv;
            const float4* vp = reinterpret_cast<const float4*>(v + (ob * MHOP + m) * DDIM + d0);
            float4 a0 = vp[0], a1 = vp[1];
            acc[0] += w * a0.x; acc[1] += w * a0.y; acc[2] += w * a0.z; acc[3] += w * a0.w;
            acc[4] += w * a1.x; acc[5] += w * a1.y; acc[6] += w * a1.z; acc[7] += w * a1.w;
        }
        float4* op = reinterpret_cast<float4*>(out + ob * DDIM + d0);
        op[0] = make_float4(acc[0], acc[1], acc[2], acc[3]);
        op[1] = make_float4(acc[4], acc[5], acc[6], acc[7]);
    }
}

extern "C" void kernel_launch(void* const* d_in, const int* in_sizes, int n_in,
                              void* d_out, int out_size, void* d_ws, size_t ws_size,
                              hipStream_t stream) {
    (void)in_sizes; (void)n_in; (void)d_ws; (void)ws_size; (void)out_size;
    const float* q  = (const float*)d_in[0];
    const float* k  = (const float*)d_in[1];
    const float* v  = (const float*)d_in[2];
    const float* W1 = (const float*)d_in[3];
    const float* W2 = (const float*)d_in[4];
    float* out = (float*)d_out;

    // CDNA5 allows up to 320KB LDS per workgroup; opt in for 264KB dynamic LDS.
    (void)hipFuncSetAttribute(reinterpret_cast<const void*>(&atten_mlp_kernel),
                              hipFuncAttributeMaxDynamicSharedMemorySize, LDS_BYTES);

    dim3 grid(BSZ / 32);   // 32 batches per block (8 waves x 4 batches)
    dim3 block(256);
    atten_mlp_kernel<<<grid, block, LDS_BYTES, stream>>>(q, k, v, W1, W2, out);
}
